// LLaMAAttention_30064771072656
// MI455X (gfx1250) — matrix operand order repaired and checked by reading, not compiled, not run on hardware
//
#include <hip/hip_runtime.h>
#include <hip/hip_bf16.h>
#include <math.h>

typedef __bf16 bf16;
typedef __attribute__((ext_vector_type(16))) __bf16 v16bf;
typedef __attribute__((ext_vector_type(8)))  float  v8f;
typedef __attribute__((ext_vector_type(4)))  int    v4i;

#define B_   2
#define S_   2048
#define D_   2048
#define H_   16
#define HD_  128
#define BS_  (B_ * S_)
#define ROWSTRIDE_ (H_ * HD_)   // 2048

// ---- async memory->LDS path (gfx1250 GLOBAL_LOAD_ASYNC_TO_LDS_B128) ----
#if defined(__has_builtin)
#if __has_builtin(__builtin_amdgcn_global_load_async_to_lds_b128)
#define USE_ASYNC_LDS 1
#endif
#endif
#ifndef USE_ASYNC_LDS
#define USE_ASYNC_LDS 0
#endif

#if USE_ASYNC_LDS
__device__ __forceinline__ void async_b128(const bf16* g, bf16* l) {
  // per-lane: LDS[l] = MEM[g], 16 bytes; tracked by ASYNCcnt
  __builtin_amdgcn_global_load_async_to_lds_b128((v4i*)g, (v4i*)l, 0, 0);
}
#define WAIT_ASYNC_0() asm volatile("s_wait_asynccnt 0x0" ::: "memory")
#define WAIT_ASYNC_4() asm volatile("s_wait_asynccnt 0x4" ::: "memory")
#else
#define WAIT_ASYNC_0()
#define WAIT_ASYNC_4()
#endif

__device__ __forceinline__ v8f zero8() {
  v8f z;
#pragma unroll
  for (int i = 0; i < 8; ++i) z[i] = 0.0f;
  return z;
}

__device__ __forceinline__ v8f wmma_bf16f32(v16bf a, v16bf b, v8f c) {
  // 8 args: (neg_a, A, neg_b, B, c_mod, C, reuse_a, reuse_b)
  return __builtin_amdgcn_wmma_f32_16x16x32_bf16(false, a, false, b, (short)0, c,
                                                 false, false);
}

// ---- fragment loaders (k = 16*half + j per lane for both A and B, wave32) ----
// Both loaders are fully contiguous per lane -> 2x ds_load_b128 each.

// A matrix 16x32 bf16 tile at (row0, k0) of row-major LDS buffer, leading dim ld.
__device__ __forceinline__ v16bf load_a_frag(const bf16* s, int row0, int ld, int k0) {
  const int lane = threadIdx.x & 31;
  const int half = lane >> 4;
  const int r    = lane & 15;
  const bf16* base = s + (row0 + r) * ld + k0;
  v16bf a;
#pragma unroll
  for (int j = 0; j < 8; ++j) a[j]     = base[half * 16 + j];
#pragma unroll
  for (int j = 0; j < 8; ++j) a[8 + j] = base[half * 16 + 8 + j];
  return a;
}

// B fragment (32x16, elem j -> K = half*16 + j, N = lane&15) taken from a
// TRANSPOSED store X[n][k] (row-major, leading dim ld): contiguous in j.
__device__ __forceinline__ v16bf load_bT_frag(const bf16* s, int n0, int ld, int k0) {
  const int lane = threadIdx.x & 31;
  const int half = lane >> 4;
  const int n    = lane & 15;
  const bf16* base = s + (n0 + n) * ld + k0 + half * 16;
  v16bf b;
#pragma unroll
  for (int j = 0; j < 16; ++j) b[j] = base[j];
  return b;
}

// ---- stage 1a: fp32 -> bf16 cast (row-major preserved) ----
__global__ void cast_f32_bf16(const float* __restrict__ in, bf16* __restrict__ out, int n4) {
  int i = blockIdx.x * blockDim.x + threadIdx.x;
  if (i >= n4) return;
  float4 v = reinterpret_cast<const float4*>(in)[i];
  bf16* o = out + 4 * (size_t)i;
  o[0] = (bf16)v.x; o[1] = (bf16)v.y; o[2] = (bf16)v.z; o[3] = (bf16)v.w;
}

// ---- stage 1b: fp32 -> bf16 cast WITH transpose: out[n][k] = in[k][n] ----
__global__ __launch_bounds__(256) void cast_transpose_f32_bf16(
    const float* __restrict__ in, bf16* __restrict__ out, int rows, int cols) {
  __shared__ bf16 tile[32][33];
  const int c0 = blockIdx.x * 32, r0 = blockIdx.y * 32;
  const int tx = threadIdx.x & 31, ty = threadIdx.x >> 5;  // ty: 0..7
#pragma unroll
  for (int i = 0; i < 4; ++i) {
    int r = ty + i * 8;
    tile[r][tx] = (bf16)in[(size_t)(r0 + r) * cols + c0 + tx];
  }
  __syncthreads();
#pragma unroll
  for (int i = 0; i < 4; ++i) {
    int r = ty + i * 8;
    out[(size_t)(c0 + r) * rows + r0 + tx] = tile[tx][r];
  }
}

// ---- stage 1c: bf16 transpose of V per (b,h): Vt[(b*H+h)*HD+d][s] = V[b,s,h,d] ----
__global__ __launch_bounds__(256) void transpose_v_bf16(const bf16* __restrict__ V,
                                                        bf16* __restrict__ Vt) {
  __shared__ bf16 tile[32][33];
  const int d0 = blockIdx.x * 32, s0 = blockIdx.y * 32, bh = blockIdx.z;
  const int b = bh >> 4, h = bh & (H_ - 1);
  const int tx = threadIdx.x & 31, ty = threadIdx.x >> 5;
#pragma unroll
  for (int i = 0; i < 4; ++i) {
    int s = s0 + ty + i * 8;
    tile[ty + i * 8][tx] = V[(size_t)(b * S_ + s) * ROWSTRIDE_ + h * HD_ + d0 + tx];
  }
  __syncthreads();
#pragma unroll
  for (int i = 0; i < 4; ++i) {
    int d = ty + i * 8;
    Vt[(size_t)(bh * HD_ + d0 + d) * S_ + s0 + tx] = tile[tx][d];
  }
}

// ---- stage 2/5: tiled bf16 WMMA GEMM, C = A(MxK) * Bt(NxK)^T ----
// 128x128 tile, BK=32, 256 threads (8 waves), double-buffered LDS with
// async memory->LDS staging pipelined against WMMA compute.
template <typename OUT>
__global__ __launch_bounds__(256) void gemm_bf16t_128(const bf16* __restrict__ A,
                                                      const bf16* __restrict__ Bt,
                                                      OUT* __restrict__ C,
                                                      int M, int N, int K) {
  __shared__ bf16 As[2][128 * 32];
  __shared__ bf16 BsT[2][128 * 32];  // [n][k]
  const int row0 = blockIdx.y * 128;
  const int col0 = blockIdx.x * 128;
  const int tid  = threadIdx.x;
  const int w    = tid >> 5;
  const int wr   = w & 3;    // 4 row groups of 32
  const int wc   = w >> 2;   // 2 col groups of 64
  const int sr   = tid >> 1, scs = (tid & 1) * 16;  // staging: 128 rows x 32 k

  v8f acc[2][4];
#pragma unroll
  for (int mi = 0; mi < 2; ++mi)
#pragma unroll
    for (int ni = 0; ni < 4; ++ni) acc[mi][ni] = zero8();

  const bf16* ga = A + (size_t)(row0 + sr) * K + scs;
  const bf16* gb = Bt + (size_t)(col0 + sr) * K + scs;
  bf16* la = &As[0][sr * 32 + scs];
  bf16* lb = &BsT[0][sr * 32 + scs];
  const int lbuf = 128 * 32;  // element stride between buffers

  // per-thread tile stage: 2x16B for A, 2x16B for B (4 async ops)
  auto stage = [&](int buf, int k0) {
#if USE_ASYNC_LDS
    async_b128(ga + k0,     la + buf * lbuf);
    async_b128(ga + k0 + 8, la + buf * lbuf + 8);
    async_b128(gb + k0,     lb + buf * lbuf);
    async_b128(gb + k0 + 8, lb + buf * lbuf + 8);
#else
    const uint4* sa = reinterpret_cast<const uint4*>(ga + k0);
    const uint4* sb = reinterpret_cast<const uint4*>(gb + k0);
    uint4* da = reinterpret_cast<uint4*>(la + buf * lbuf);
    uint4* db = reinterpret_cast<uint4*>(lb + buf * lbuf);
    da[0] = sa[0]; da[1] = sa[1];
    db[0] = sb[0]; db[1] = sb[1];
#endif
  };

  const int NT = K >> 5;
  stage(0, 0);
  for (int kt = 0; kt < NT; ++kt) {
    const int cur = kt & 1;
    if (kt + 1 < NT) {
      stage(cur ^ 1, (kt + 1) << 5);  // overlap next tile's loads with compute
      WAIT_ASYNC_4();                 // oldest 4 (current tile) complete, in order
    } else {
      WAIT_ASYNC_0();
    }
    __syncthreads();

    const bf16* as = As[cur];
    const bf16* bs = BsT[cur];
    v16bf af0 = load_a_frag(as, 32 * wr,      32, 0);
    v16bf af1 = load_a_frag(as, 32 * wr + 16, 32, 0);
#pragma unroll
    for (int ni = 0; ni < 4; ++ni) {
      v16bf bfrag = load_bT_frag(bs, 64 * wc + 16 * ni, 32, 0);
      acc[0][ni] = wmma_bf16f32(af0, bfrag, acc[0][ni]);
      acc[1][ni] = wmma_bf16f32(af1, bfrag, acc[1][ni]);
    }
    __syncthreads();  // all waves done with 'cur' before it is restaged
  }

  const int lane = tid & 31, half = lane >> 4, n = lane & 15;
#pragma unroll
  for (int mi = 0; mi < 2; ++mi)
#pragma unroll
    for (int ni = 0; ni < 4; ++ni)
#pragma unroll
      for (int i = 0; i < 8; ++i) {
        int rr = row0 + 32 * wr + 16 * mi + half * 8 + i;
        int cc = col0 + 64 * wc + 16 * ni + n;
        C[(size_t)rr * N + cc] = (OUT)acc[mi][ni][i];
      }
}

// ---- stage 3: RoPE on Q and K (in-place, bf16) ----
__global__ void rope_kernel(bf16* __restrict__ Q, bf16* __restrict__ Kd) {
  int idx = blockIdx.x * blockDim.x + threadIdx.x;   // over BS_*H_*64 pairs
  int p   = idx & 63;
  int h   = (idx >> 6) & (H_ - 1);
  int row = idx >> 10;                               // 0..BS_-1
  int s   = row & (S_ - 1);
  float freq = __expf(-(float)(2 * p) * 0.0078125f * __logf(10000.0f)); // 10000^(-2p/128)
  float ang = (float)s * freq;
  float sn, cs;
  __sincosf(ang, &sn, &cs);
  size_t off = (size_t)row * ROWSTRIDE_ + h * HD_ + 2 * p;
  float q1 = (float)Q[off], q2 = (float)Q[off + 1];
  Q[off]     = (bf16)(q1 * cs - q2 * sn);
  Q[off + 1] = (bf16)(q1 * sn + q2 * cs);
  float k1 = (float)Kd[off], k2 = (float)Kd[off + 1];
  Kd[off]     = (bf16)(k1 * cs - k2 * sn);
  Kd[off + 1] = (bf16)(k1 * sn + k2 * cs);
}

// ---- stage 4: causal flash attention, 128 queries per block, 64-key blocks ----
// Q, K in (b,s,h,d) bf16; V pre-transposed: Vt[(b*H+h)*HD+d][s].
__global__ __launch_bounds__(256) void attn_kernel(const bf16* __restrict__ Q,
                                                   const bf16* __restrict__ Kg,
                                                   const bf16* __restrict__ Vt,
                                                   bf16* __restrict__ CTX) {
  __shared__ bf16 Qs[128 * 128];  // 32 KB
  __shared__ bf16 Ks[64 * 128];   // 16 KB; reused as P (8 waves x 16x64) after scores
  __shared__ bf16 VsT[128 * 64];  // 16 KB: [d][key]
  const int qt = blockIdx.x, h = blockIdx.y, b = blockIdx.z;
  const int bh = b * H_ + h;
  const int q0  = qt * 128;
  const int tid = threadIdx.x;
  const int w   = tid >> 5;
  const int lane = tid & 31, half = lane >> 4, n = lane & 15;

  { // stage Q tile 128x128
    int r = tid >> 1, cs = (tid & 1) * 64;
    const bf16* src = Q + (size_t)(b * S_ + q0 + r) * ROWSTRIDE_ + h * HD_ + cs;
    bf16* dst = Qs + r * 128 + cs;
#if USE_ASYNC_LDS
#pragma unroll
    for (int j = 0; j < 8; ++j) async_b128(src + 8 * j, dst + 8 * j);
#else
#pragma unroll
    for (int j = 0; j < 8; ++j)
      reinterpret_cast<uint4*>(dst)[j] = reinterpret_cast<const uint4*>(src)[j];
#endif
  }

  v8f ctx[8];
  float m[8], l[8];
#pragma unroll
  for (int i = 0; i < 8; ++i) { m[i] = -1e30f; l[i] = 0.0f; }
#pragma unroll
  for (int nf = 0; nf < 8; ++nf) ctx[nf] = zero8();

  const int nkb = (q0 + 128) >> 6;           // causal: key blocks with key0 <= q0+127
  const float scale = 0.08838834764831845f;  // 1/sqrt(128)
  const int qrow_base = q0 + 16 * w + half * 8;

  // staging geometry
  const int kr = tid >> 2, kcs = (tid & 3) * 32;  // K: 64 rows x 128 d
  const int vr = tid >> 1, vcs = (tid & 1) * 32;  // Vt: 128 d-rows x 64 keys

  for (int kb = 0; kb < nkb; ++kb) {
    const int key0 = kb * 64;
    { // stage K (row-major) and Vt (transposed) blocks
      const bf16* sk = Kg + (size_t)(b * S_ + key0 + kr) * ROWSTRIDE_ + h * HD_ + kcs;
      bf16* dk = Ks + kr * 128 + kcs;
      const bf16* sv = Vt + (size_t)(bh * HD_ + vr) * S_ + key0 + vcs;
      bf16* dv = VsT + vr * 64 + vcs;
#if USE_ASYNC_LDS
#pragma unroll
      for (int j = 0; j < 4; ++j) async_b128(sk + 8 * j, dk + 8 * j);
#pragma unroll
      for (int j = 0; j < 4; ++j) async_b128(sv + 8 * j, dv + 8 * j);
#else
#pragma unroll
      for (int j = 0; j < 4; ++j) {
        reinterpret_cast<uint4*>(dk)[j] = reinterpret_cast<const uint4*>(sk)[j];
        reinterpret_cast<uint4*>(dv)[j] = reinterpret_cast<const uint4*>(sv)[j];
      }
#endif
    }
    if (kb + 1 < nkb) {  // prefetch next key block (global_prefetch_b8)
      __builtin_prefetch(Kg + (size_t)(b * S_ + key0 + 64 + kr) * ROWSTRIDE_ + h * HD_ + kcs, 0, 3);
      __builtin_prefetch(Vt + (size_t)(bh * HD_ + vr) * S_ + key0 + 64 + vcs, 0, 3);
    }
    WAIT_ASYNC_0();
    __syncthreads();

    // scores: this wave's 16 q rows x 64 keys = 4 frags, K-dim = HD (4 steps of 32)
    v8f sf[4];
#pragma unroll
    for (int c = 0; c < 4; ++c) sf[c] = zero8();
#pragma unroll
    for (int ds = 0; ds < 4; ++ds) {
      v16bf aq = load_a_frag(Qs, 16 * w, 128, 32 * ds);
#pragma unroll
      for (int c = 0; c < 4; ++c) {
        v16bf bk = load_bT_frag(Ks, 16 * c, 128, 32 * ds);  // K rows are "n", contiguous d
        sf[c] = wmma_bf16f32(aq, bk, sf[c]);
      }
    }

    // scale + causal mask
    if (key0 + 63 > q0) {
#pragma unroll
      for (int c = 0; c < 4; ++c) {
        const int key = key0 + 16 * c + n;
#pragma unroll
        for (int i = 0; i < 8; ++i) {
          float v = sf[c][i] * scale;
          sf[c][i] = (key > qrow_base + i) ? -1e30f : v;
        }
      }
    } else {
#pragma unroll
      for (int c = 0; c < 4; ++c)
#pragma unroll
        for (int i = 0; i < 8; ++i) sf[c][i] *= scale;
    }

    // online softmax: a full score row lives in one 16-lane half-group
#pragma unroll
    for (int i = 0; i < 8; ++i) {
      float rm = fmaxf(fmaxf(sf[0][i], sf[1][i]), fmaxf(sf[2][i], sf[3][i]));
#pragma unroll
      for (int d = 1; d < 16; d <<= 1) rm = fmaxf(rm, __shfl_xor(rm, d, 32));
      float mn = fmaxf(m[i], rm);
      float al = __expf(m[i] - mn);
      float ls = 0.0f;
#pragma unroll
      for (int c = 0; c < 4; ++c) {
        float p = __expf(sf[c][i] - mn);
        sf[c][i] = p;
        ls += p;
      }
#pragma unroll
      for (int d = 1; d < 16; d <<= 1) ls += __shfl_xor(ls, d, 32);
      l[i] = l[i] * al + ls;
      m[i] = mn;
#pragma unroll
      for (int nf = 0; nf < 8; ++nf) ctx[nf][i] = ctx[nf][i] * al;
    }

    __syncthreads();  // all waves done reading Ks -> safe to overwrite with P

    // write P (C-layout -> LDS row-major 16x64) to transpose into A-layout
    bf16* Pw = Ks + w * (16 * 64);
#pragma unroll
    for (int c = 0; c < 4; ++c)
#pragma unroll
      for (int i = 0; i < 8; ++i)
        Pw[(half * 8 + i) * 64 + 16 * c + n] = (bf16)sf[c][i];
    asm volatile("s_wait_dscnt 0" ::: "memory");  // wave-local LDS RAW

    // ctx += P(16x64) @ V(64x128): 2 k-steps x 8 col frags, B from VsT (contiguous)
#pragma unroll
    for (int ks = 0; ks < 2; ++ks) {
      v16bf ap = load_a_frag(Pw, 0, 64, 32 * ks);
#pragma unroll
      for (int nf = 0; nf < 8; ++nf) {
        v16bf bv = load_bT_frag(VsT, 16 * nf, 64, 32 * ks);
        ctx[nf] = wmma_bf16f32(ap, bv, ctx[nf]);
      }
    }
    __syncthreads();  // before next iteration overwrites Ks/VsT
  }

  // normalize and write ctx (bf16) back in (b, s, h*HD + d) layout
#pragma unroll
  for (int i = 0; i < 8; ++i) l[i] = 1.0f / l[i];
#pragma unroll
  for (int nf = 0; nf < 8; ++nf)
#pragma unroll
    for (int i = 0; i < 8; ++i) {
      int rr = q0 + 16 * w + half * 8 + i;
      int cc = 16 * nf + n;
      CTX[(size_t)(b * S_ + rr) * ROWSTRIDE_ + h * HD_ + cc] = (bf16)(ctx[nf][i] * l[i]);
    }
}

extern "C" void kernel_launch(void* const* d_in, const int* in_sizes, int n_in,
                              void* d_out, int out_size, void* d_ws, size_t ws_size,
                              hipStream_t stream) {
  (void)in_sizes; (void)n_in; (void)out_size; (void)ws_size;
  const float* x  = (const float*)d_in[0];
  const float* Wq = (const float*)d_in[1];
  const float* Wk = (const float*)d_in[2];
  const float* Wv = (const float*)d_in[3];
  const float* Wo = (const float*)d_in[4];
  float* out = (float*)d_out;

  bf16* ws = (bf16*)d_ws;
  const size_t nx = (size_t)BS_ * D_;        // 8,388,608
  const size_t nw = (size_t)D_ * ROWSTRIDE_; // 4,194,304
  size_t off = 0;
  bf16* xb   = ws + off; off += nx;
  bf16* WqT  = ws + off; off += nw;   // transposed bf16 weights (N x K)
  bf16* WkT  = ws + off; off += nw;
  bf16* WvT  = ws + off; off += nw;
  bf16* WoT  = ws + off; off += nw;
  bf16* Qb   = ws + off; off += nx;
  bf16* Kb   = ws + off; off += nx;
  bf16* Vb   = ws + off; off += nx;
  bf16* Vtb  = ws + off; off += nx;   // V transposed per (b,h): [bh*HD+d][s]
  bf16* Cb   = ws + off; off += nx;   // ~134 MB total

  cast_f32_bf16<<<(int)(nx / 4 / 256), 256, 0, stream>>>(x, xb, (int)(nx / 4));
  dim3 tgrid(D_ / 32, D_ / 32);  // (64, 64)
  cast_transpose_f32_bf16<<<tgrid, 256, 0, stream>>>(Wq, WqT, D_, ROWSTRIDE_);
  cast_transpose_f32_bf16<<<tgrid, 256, 0, stream>>>(Wk, WkT, D_, ROWSTRIDE_);
  cast_transpose_f32_bf16<<<tgrid, 256, 0, stream>>>(Wv, WvT, D_, ROWSTRIDE_);
  cast_transpose_f32_bf16<<<tgrid, 256, 0, stream>>>(Wo, WoT, ROWSTRIDE_, D_);

  dim3 ggrid(ROWSTRIDE_ / 128, BS_ / 128);  // (16, 32)
  gemm_bf16t_128<bf16><<<ggrid, 256, 0, stream>>>(xb, WqT, Qb, BS_, ROWSTRIDE_, D_);
  gemm_bf16t_128<bf16><<<ggrid, 256, 0, stream>>>(xb, WkT, Kb, BS_, ROWSTRIDE_, D_);
  gemm_bf16t_128<bf16><<<ggrid, 256, 0, stream>>>(xb, WvT, Vb, BS_, ROWSTRIDE_, D_);

  rope_kernel<<<(BS_ * H_ * 64) / 256, 256, 0, stream>>>(Qb, Kb);

  transpose_v_bf16<<<dim3(HD_ / 32, S_ / 32, B_ * H_), 256, 0, stream>>>(Vb, Vtb);

  attn_kernel<<<dim3(S_ / 128, H_, B_), 256, 0, stream>>>(Qb, Kb, Vtb, Cb);

  gemm_bf16t_128<float><<<dim3(D_ / 128, BS_ / 128), 256, 0, stream>>>(
      Cb, WoT, out, BS_, D_, ROWSTRIDE_);
}
